// NeocortexBlock_83030307766297
// MI455X (gfx1250) — matrix operand order repaired and checked by reading, not compiled, not run on hardware
//
#include <hip/hip_runtime.h>
#include <hip/hip_bf16.h>
#include <math.h>

// ---------------------------------------------------------------------------
// Shapes (fixed by the reference)
// ---------------------------------------------------------------------------
#define NHEAD 16
#define DHEAD 128
#define HIDSZ 512
#define DTOT  2048        // NHEAD * DHEAD
#define TM    128         // tokens per workgroup tile (8 row tiles of 16)
#define LSTRIDE 520       // f16 row stride in LDS (512 + 8 pad, 16B aligned rows)

typedef __attribute__((ext_vector_type(16))) _Float16 v16h;
typedef __attribute__((ext_vector_type(8)))  _Float16 v8h;
typedef __attribute__((ext_vector_type(8)))  float    v8f;

// ---------------------------------------------------------------------------
// Helpers
// ---------------------------------------------------------------------------
__device__ __forceinline__ v16h ld_frag_split(const _Float16* p0, const _Float16* p1) {
  v16h r;
  ((v8h*)&r)[0] = *(const v8h*)p0;   // elements 0..7
  ((v8h*)&r)[1] = *(const v8h*)p1;   // elements 8..15
  return r;
}

__device__ __forceinline__ float gelu_exact(float v) {
  return 0.5f * v * (1.0f + erff(v * 0.70710678118654752f));
}

// ---------------------------------------------------------------------------
// Kernel 1: weight convert + transpose  w[h][k][n] (f32) -> wT[h][n][k] (f16)
// ---------------------------------------------------------------------------
__global__ __launch_bounds__(256) void wconv_kernel(const float* __restrict__ src,
                                                    _Float16* __restrict__ dst,
                                                    int K, int N) {
  const long long KN = (long long)K * N;
  const long long total = (long long)NHEAD * KN;
  for (long long idx = (long long)blockIdx.x * blockDim.x + threadIdx.x;
       idx < total; idx += (long long)gridDim.x * blockDim.x) {
    long long h = idx / KN;
    long long r = idx - h * KN;
    int k = (int)(r / N);
    int n = (int)(r - (long long)k * N);
    dst[h * KN + (long long)n * K + k] = (_Float16)src[idx];
  }
}

// ---------------------------------------------------------------------------
// Kernel 2: per-token inverse RMS  (eps = FLT_EPSILON, matches nn.RMSNorm)
// ---------------------------------------------------------------------------
__global__ __launch_bounds__(256) void rms_kernel(const float* __restrict__ x,
                                                  float* __restrict__ scale) {
  __shared__ float red[8];
  const size_t token = blockIdx.x;
  const float* xr = x + token * (size_t)DTOT;
  float s = 0.0f;
  for (int i = threadIdx.x; i < DTOT; i += 256) {
    float v = xr[i];
    s += v * v;
  }
  for (int off = 16; off > 0; off >>= 1) s += __shfl_down(s, off, 32);
  if ((threadIdx.x & 31) == 0) red[threadIdx.x >> 5] = s;
  __syncthreads();
  if (threadIdx.x == 0) {
    float t = 0.0f;
    #pragma unroll
    for (int i = 0; i < 8; ++i) t += red[i];
    scale[token] = rsqrtf(t * (1.0f / (float)DTOT) + 1.1920929e-07f);
  }
}

// ---------------------------------------------------------------------------
// One MLP layer with N = 512 outputs (layers 0,1,2), GELU on output.
// in / out are LDS activation buffers (row-major, stride LSTRIDE f16).
// wTh points at this head's transposed f16 weights: [512][K].
// 8 waves: wave w owns column tiles {w*4 .. w*4+3} (2 per pass), ALL 8 row
// tiles -> each B fragment is loaded once per workgroup and reused 8x.
// ---------------------------------------------------------------------------
template <int K, bool DOGELU>
__device__ __forceinline__ void mlp_layer512(const _Float16* __restrict__ in,
                                             _Float16* __restrict__ out,
                                             const _Float16* __restrict__ wTh,
                                             int wave, int lane) {
  const int lm = lane & 15;          // matrix row (A) / column (B,C)
  const int hi = (lane >> 4) & 1;    // lane-half select
  for (int pass = 0; pass < 2; ++pass) {
    const int c0 = wave * 4 + pass * 2;     // column tile base (of 32)
    v8f acc[8][2];
    #pragma unroll
    for (int r = 0; r < 8; ++r)
      #pragma unroll
      for (int cc = 0; cc < 2; ++cc)
        acc[r][cc] = (v8f)(0.0f);

    for (int kt = 0; kt < K / 32; ++kt) {
      const int kb = kt * 32;
      // B fragments: contiguous K in transposed weights (32B aligned)
      v16h b[2];
      #pragma unroll
      for (int cc = 0; cc < 2; ++cc) {
        const int n = (c0 + cc) * 16 + lm;
        const _Float16* bp = wTh + (size_t)n * K + kb + hi * 16;
        b[cc] = *(const v16h*)bp;
      }
      // A fragments from LDS: K {0..15} (lanes 0-15) / {16..31}
      #pragma unroll
      for (int r = 0; r < 8; ++r) {
        const _Float16* ap = in + (r * 16 + lm) * LSTRIDE + kb + hi * 16;
        const v16h a = ld_frag_split(ap, ap + 8);
        acc[r][0] = __builtin_amdgcn_wmma_f32_16x16x32_f16(
            false, a, false, b[0], (short)0, acc[r][0], false, false);
        acc[r][1] = __builtin_amdgcn_wmma_f32_16x16x32_f16(
            false, a, false, b[1], (short)0, acc[r][1], false, false);
      }
    }
    // GELU + f16 write-back following the C/D VGPR layout
    #pragma unroll
    for (int r = 0; r < 8; ++r) {
      #pragma unroll
      for (int cc = 0; cc < 2; ++cc) {
        const int col = (c0 + cc) * 16 + lm;
        #pragma unroll
        for (int v = 0; v < 8; ++v) {
          float f = acc[r][cc][v];
          if (DOGELU) f = gelu_exact(f);
          const int row = r * 16 + v + hi * 8;
          out[row * LSTRIDE + col] = (_Float16)f;
        }
      }
    }
  }
}

// ---------------------------------------------------------------------------
// Final layer: K = 512, N = 128, no GELU, fp32 store straight to d_out.
// Wave w owns column tile w (8 tiles cover N=128), all 8 row tiles.
// ---------------------------------------------------------------------------
__device__ __forceinline__ void mlp_layer_out(const _Float16* __restrict__ in,
                                              float* __restrict__ outg,
                                              const _Float16* __restrict__ wTh,
                                              int wave, int lane, int m0, int head) {
  const int lm = lane & 15;
  const int hi = (lane >> 4) & 1;
  const int c = wave;
  v8f acc[8];
  #pragma unroll
  for (int r = 0; r < 8; ++r) acc[r] = (v8f)(0.0f);

  for (int kt = 0; kt < HIDSZ / 32; ++kt) {
    const int kb = kt * 32;
    const int n = c * 16 + lm;
    const _Float16* bp = wTh + (size_t)n * HIDSZ + kb + hi * 16;
    const v16h b = *(const v16h*)bp;
    #pragma unroll
    for (int r = 0; r < 8; ++r) {
      const _Float16* ap = in + (r * 16 + lm) * LSTRIDE + kb + hi * 16;
      const v16h a = ld_frag_split(ap, ap + 8);
      acc[r] = __builtin_amdgcn_wmma_f32_16x16x32_f16(
          false, a, false, b, (short)0, acc[r], false, false);
    }
  }
  #pragma unroll
  for (int r = 0; r < 8; ++r) {
    const int col = c * 16 + lm;
    #pragma unroll
    for (int v = 0; v < 8; ++v) {
      const int row = r * 16 + v + hi * 8;
      outg[(size_t)(m0 + row) * DTOT + head * DHEAD + col] = acc[r][v];
    }
  }
}

// ---------------------------------------------------------------------------
// Kernel 3: fused per-head MLP chain. grid = (ntok/TM, NHEAD), block = 256.
// Dynamic LDS: two ping-pong activation buffers of TM x LSTRIDE f16 (~260KB).
// ---------------------------------------------------------------------------
__global__ __launch_bounds__(256) void NeocortexBlock_83030307766297_kernel(
    const float* __restrict__ x, const float* __restrict__ g,
    const float* __restrict__ scale,
    const _Float16* __restrict__ w0T, const _Float16* __restrict__ w1T,
    const _Float16* __restrict__ w2T, const _Float16* __restrict__ w3T,
    float* __restrict__ out) {
  extern __shared__ __align__(16) _Float16 smem[];
  _Float16* buf0 = smem;
  _Float16* buf1 = smem + TM * LSTRIDE;

  const int m0   = blockIdx.x * TM;
  const int head = blockIdx.y;
  const int tid  = threadIdx.x;
  const int wave = tid >> 5;
  const int lane = tid & 31;

  // Stage normalized input slice (TM x 128) as f16 into LDS (coalesced).
  for (int i = tid; i < TM * DHEAD; i += 256) {
    const int row = i >> 7;
    const int d   = i & 127;
    const float v = x[(size_t)(m0 + row) * DTOT + head * DHEAD + d] *
                    scale[m0 + row] * g[head * DHEAD + d];
    buf0[row * LSTRIDE + d] = (_Float16)v;
  }
  __syncthreads();

  mlp_layer512<DHEAD, true>(buf0, buf1, w0T + (size_t)head * DHEAD * HIDSZ, wave, lane);
  __syncthreads();
  mlp_layer512<HIDSZ, true>(buf1, buf0, w1T + (size_t)head * HIDSZ * HIDSZ, wave, lane);
  __syncthreads();
  mlp_layer512<HIDSZ, true>(buf0, buf1, w2T + (size_t)head * HIDSZ * HIDSZ, wave, lane);
  __syncthreads();
  mlp_layer_out(buf1, out, w3T + (size_t)head * HIDSZ * DHEAD, wave, lane, m0, head);
}

// ---------------------------------------------------------------------------
// Host launcher
// ---------------------------------------------------------------------------
extern "C" void kernel_launch(void* const* d_in, const int* in_sizes, int n_in,
                              void* d_out, int out_size, void* d_ws, size_t ws_size,
                              hipStream_t stream) {
  const float* x  = (const float*)d_in[0];
  const float* g  = (const float*)d_in[1];
  const float* w0 = (const float*)d_in[2];
  const float* w1 = (const float*)d_in[3];
  const float* w2 = (const float*)d_in[4];
  const float* w3 = (const float*)d_in[5];
  float* out = (float*)d_out;

  const int ntok = in_sizes[0] / DTOT;   // 16384

  // Workspace layout
  char* ws = (char*)d_ws;
  float* scale = (float*)ws;
  size_t off = (size_t)ntok * sizeof(float);
  off = (off + 255) & ~(size_t)255;
  _Float16* w0T = (_Float16*)(ws + off); off += (size_t)NHEAD * DHEAD * HIDSZ * sizeof(_Float16);
  _Float16* w1T = (_Float16*)(ws + off); off += (size_t)NHEAD * HIDSZ * HIDSZ * sizeof(_Float16);
  _Float16* w2T = (_Float16*)(ws + off); off += (size_t)NHEAD * HIDSZ * HIDSZ * sizeof(_Float16);
  _Float16* w3T = (_Float16*)(ws + off); off += (size_t)NHEAD * HIDSZ * DHEAD * sizeof(_Float16);

  // 1) weight convert + transpose (f32 -> f16, [h][n][k])
  wconv_kernel<<<2048, 256, 0, stream>>>(w0, w0T, DHEAD, HIDSZ);
  wconv_kernel<<<4096, 256, 0, stream>>>(w1, w1T, HIDSZ, HIDSZ);
  wconv_kernel<<<4096, 256, 0, stream>>>(w2, w2T, HIDSZ, HIDSZ);
  wconv_kernel<<<2048, 256, 0, stream>>>(w3, w3T, HIDSZ, DHEAD);

  // 2) per-token inverse RMS
  rms_kernel<<<ntok, 256, 0, stream>>>(x, scale);

  // 3) fused per-head MLP chain on WMMA
  const size_t smem_bytes = (size_t)2 * TM * LSTRIDE * sizeof(_Float16); // 266,240 B
  (void)hipFuncSetAttribute((const void*)NeocortexBlock_83030307766297_kernel,
                            hipFuncAttributeMaxDynamicSharedMemorySize,
                            (int)smem_bytes);
  dim3 grid(ntok / TM, NHEAD);
  NeocortexBlock_83030307766297_kernel<<<grid, 256, smem_bytes, stream>>>(
      x, g, scale, w0T, w1T, w2T, w3T, out);
}
